// GNNCapModel_37168646979921
// MI455X (gfx1250) — matrix operand order repaired and checked by reading, not compiled, not run on hardware
//
#include <hip/hip_runtime.h>
#include <hip/hip_bf16.h>

typedef __bf16 bf16;
typedef __attribute__((ext_vector_type(16))) __bf16 v16bf;
typedef __attribute__((ext_vector_type(8)))  __bf16 v8bf;
typedef __attribute__((ext_vector_type(8)))  float  v8f;

#define TPB   256
#define NWAVE 8
#define MAXK  288   // largest concatenated (padded) K: 128+128+32

struct Seg { const bf16* A; const int* idx; const bf16* Wt; int K; };
struct GemmP {
  Seg seg[3];
  int nseg, M, Np, Nact, relu, ldout, ldS;
  const float* bias;
  bf16* outB;          // bf16 dense output mode (if non-null)
  float* outS;         // fp32 atomic-scatter mode otherwise
  const int* sidx;
};

// ---------------- WMMA GEMM: C[M,Np] = concat_segs(A[idx]) * Wt^T + bias ----
__launch_bounds__(TPB)
__global__ void gemm_wmma_k(GemmP p) {
  __shared__ __align__(32) bf16 ldsA[16 * MAXK];
  int Ktot = 0;
  for (int s = 0; s < p.nseg; ++s) Ktot += p.seg[s].K;
  const int K2 = Ktot >> 1;
  const int m0 = blockIdx.x * 16;
  const int tid = threadIdx.x;

  // cooperative gather of 16 (padded) A rows into LDS, dword-granular
  unsigned* ldsU = (unsigned*)ldsA;
  for (int i = tid; i < 16 * K2; i += TPB) {
    int r = i / K2, c = (i - r * K2) << 1;
    unsigned v = 0u;
    int m = m0 + r;
    if (m < p.M) {
      int s = 0, base = 0;
      while (c >= base + p.seg[s].K) { base += p.seg[s].K; ++s; }
      int g = p.seg[s].idx ? p.seg[s].idx[m] : m;
      const unsigned* src = (const unsigned*)(p.seg[s].A + (size_t)g * p.seg[s].K);
      v = src[(c - base) >> 1];
    }
    ldsU[i] = v;
  }
  __syncthreads();

  const int wave = tid >> 5, lane = tid & 31;
  const int lh = lane >> 4, lm = lane & 15;
  const int ntiles = p.Np >> 4;

  for (int nt = wave; nt < ntiles; nt += NWAVE) {
    const int n = nt * 16 + lm;                 // C/D: lane%16 -> N
    float bv = (p.bias && n < p.Nact) ? p.bias[n] : 0.0f;
    v8f acc = { bv, bv, bv, bv, bv, bv, bv, bv };
    int base = 0;
    for (int s = 0; s < p.nseg; ++s) {
      const int Ks = p.seg[s].K;
      const bf16* wrow = p.seg[s].Wt + (size_t)n * Ks;   // Wt is [Np][Kp]
      for (int kk = 0; kk < Ks; kk += 32) {
        // A 16x32 bf16 fragment: lane<16 K={0..7,16..23}, lane>=16 +8
        const bf16* ap = &ldsA[lm * Ktot + base + kk + lh * 8];
        v8bf a0 = *(const v8bf*)ap;
        v8bf a1 = *(const v8bf*)(ap + 16);
        v16bf a;
        #pragma unroll
        for (int j = 0; j < 8; ++j) { a[j] = a0[j]; a[j + 8] = a1[j]; }
        // B 32x16 fragment: lane n=lane%16, K={0..7,16..23}, lane>=16 +8
        const bf16* bp = wrow + kk + lh * 8;
        v8bf b0 = *(const v8bf*)bp;
        v8bf b1 = *(const v8bf*)(bp + 16);
        v16bf b;
        #pragma unroll
        for (int j = 0; j < 8; ++j) { b[j] = b0[j]; b[j + 8] = b1[j]; }
        acc = __builtin_amdgcn_wmma_f32_16x16x32_bf16(
                false, a, false, b, (short)0, acc, false, false);
      }
      base += Ks;
    }
    if (p.outB) {
      #pragma unroll
      for (int r = 0; r < 8; ++r) {
        int m = m0 + r + lh * 8;                // C/D: VGPR r -> M, +8 for hi lanes
        if (m < p.M) {
          float v = acc[r];
          if (p.relu) v = v > 0.f ? v : 0.f;
          p.outB[(size_t)m * p.ldout + n] = (bf16)v;
        }
      }
    } else if (n < p.Nact) {
      #pragma unroll
      for (int r = 0; r < 8; ++r) {
        int m = m0 + r + lh * 8;
        if (m < p.M) atomicAdd(&p.outS[(size_t)p.sidx[m] * p.ldS + n], acc[r]);
      }
    }
  }
}

// ---------------- small helper kernels -------------------------------------
__global__ void pad3_k(const float* __restrict__ x, bf16* __restrict__ o, int n) {
  int t = blockIdx.x * blockDim.x + threadIdx.x;
  if (t >= n * 32) return;
  int i = t >> 5, c = t & 31;
  o[t] = (c < 3) ? (bf16)x[i * 3 + c] : (bf16)0.0f;
}
__global__ void pad7_k(const float* __restrict__ x, bf16* __restrict__ o, int n) {
  int t = blockIdx.x * blockDim.x + threadIdx.x;
  if (t >= n * 32) return;
  int i = t >> 5, c = t & 31;
  o[t] = (c < 7) ? (bf16)x[i * 7 + c] : (bf16)0.0f;
}
__global__ void zero_k(float* p, long long n) {
  long long t = (long long)blockIdx.x * blockDim.x + threadIdx.x;
  if (t < n) p[t] = 0.0f;
}
__global__ void count_k(const int* __restrict__ col, int* cnt, int e) {
  int t = blockIdx.x * blockDim.x + threadIdx.x;
  if (t < e) atomicAdd(&cnt[col[t]], 1);
}
// nu_in1[i] = concat(x[i](3), ssum/cnt(42), zeros) padded to 64
__global__ void finish1_k(const float* __restrict__ x, const float* __restrict__ ss,
                          const int* __restrict__ cnt, bf16* __restrict__ o, int n) {
  int t = blockIdx.x * blockDim.x + threadIdx.x;
  if (t >= n * 64) return;
  int i = t >> 6, c = t & 63;
  float v = 0.f;
  if (c < 3) v = x[i * 3 + c];
  else if (c < 45) v = ss[i * 64 + (c - 3)] / fmaxf((float)cnt[i], 1.0f);
  o[t] = (bf16)v;
}
// nu_in2[i] = concat(h1[i](128), ssum2/cnt(112), zeros) padded to 256
__global__ void finish2_k(const bf16* __restrict__ h1, const float* __restrict__ ss,
                          const int* __restrict__ cnt, bf16* __restrict__ o, int n) {
  int t = blockIdx.x * blockDim.x + threadIdx.x;
  if (t >= n * 256) return;
  int i = t >> 8, c = t & 255;
  bf16 v = (bf16)0.0f;
  if (c < 128) v = h1[i * 128 + c];
  else if (c < 240) v = (bf16)(ss[i * 128 + (c - 128)] / fmaxf((float)cnt[i], 1.0f));
  o[t] = v;
}
__global__ void fc_k(const bf16* __restrict__ h, const float* __restrict__ W,
                     const float* __restrict__ b, float* __restrict__ out, int n) {
  int t = blockIdx.x * blockDim.x + threadIdx.x;
  if (t >= n) return;
  float s = b[0];
  const bf16* hp = h + (size_t)t * 288;
  for (int k = 0; k < 264; ++k) s += (float)hp[k] * W[k];
  out[t] = s;
}
// C[K,Nn] = A[aoff:,:K x J](lda) * B[boff:,:J x Nn](ldb) + (D ? D[doff:] : 0)
__global__ void mm_k(float* __restrict__ C, const float* __restrict__ A, int lda, int aoff,
                     const float* __restrict__ B, int ldb, int boff,
                     const float* __restrict__ D, int ldd, int doff,
                     int K, int J, int Nn) {
  int t = blockIdx.x * blockDim.x + threadIdx.x;
  if (t >= K * Nn) return;
  int k = t / Nn, n = t - k * Nn;
  float s = D ? D[(size_t)(doff + k) * ldd + n] : 0.0f;
  for (int j = 0; j < J; ++j)
    s += A[(size_t)(aoff + k) * lda + j] * B[(size_t)(boff + j) * ldb + n];
  C[t] = s;
}
// Wt[Np][Kp] (bf16, transposed, zero-padded) from fp32 W rows [roff, roff+kseg)
__global__ void cvt_k(const float* __restrict__ W, int ldW, int roff, int kseg, int dout,
                      bf16* __restrict__ Wt, int Kp, int Np) {
  int t = blockIdx.x * blockDim.x + threadIdx.x;
  if (t >= Np * Kp) return;
  int n = t / Kp, k = t - n * Kp;
  Wt[t] = (k < kseg && n < dout) ? (bf16)W[(size_t)(roff + k) * ldW + n] : (bf16)0.0f;
}

// ---------------- host ------------------------------------------------------
static inline Seg mkseg(const bf16* A, const int* idx, const bf16* Wt, int K) {
  Seg s; s.A = A; s.idx = idx; s.Wt = Wt; s.K = K; return s;
}
static void gemm(hipStream_t st, int M, int Np, int Nact, const float* bias,
                 Seg s0, Seg s1, Seg s2, int nseg,
                 bf16* outB, int ldout, int relu,
                 float* outS, const int* sidx, int ldS) {
  GemmP p{};
  p.seg[0] = s0; p.seg[1] = s1; p.seg[2] = s2; p.nseg = nseg;
  p.M = M; p.Np = Np; p.Nact = Nact; p.relu = relu; p.ldout = ldout; p.ldS = ldS;
  p.bias = bias; p.outB = outB; p.outS = outS; p.sidx = sidx;
  gemm_wmma_k<<<dim3((M + 15) / 16), dim3(TPB), 0, st>>>(p);
}

extern "C" void kernel_launch(void* const* d_in, const int* in_sizes, int n_in,
                              void* d_out, int out_size, void* d_ws, size_t ws_size,
                              hipStream_t stream) {
  const int N = 50000, E = 500000;
  auto fi = [&](int sz) -> int {
    for (int i = 0; i < n_in; ++i) if (in_sizes[i] == sz) return i;
    return -1;
  };
  int ix = fi(150000), ie = fi(1000000), iea = fi(3500000);
  int inb1 = fi(420), inu11 = fi(3735), inu12 = fi(5893), inu13 = fi(9088);
  int ied1 = fi(896), ieu1 = fi(49152), inb2 = fi(28672);
  int inu21 = fi(44160), inu22 = fi(14720), inu23 = fi(21120);
  int ifc = -1;
  for (int i = 0; i + 1 < n_in; ++i)
    if (in_sizes[i] == 264 && in_sizes[i + 1] == 1) { ifc = i; break; }
  if (ix < 0 || ie < 0 || iea < 0 || inb1 < 0 || inu11 < 0 || inu12 < 0 || inu13 < 0 ||
      ied1 < 0 || ieu1 < 0 || inb2 < 0 || inu21 < 0 || inu22 < 0 || inu23 < 0 || ifc < 0)
    return;

  const float* x    = (const float*)d_in[ix];
  const int*   row  = (const int*)d_in[ie];
  const int*   col  = row + E;
  const float* ea   = (const float*)d_in[iea];
  const float* Wnb1 = (const float*)d_in[inb1],  *bnb1 = (const float*)d_in[inb1 + 1];
  const float* Wn11 = (const float*)d_in[inu11], *bn11 = (const float*)d_in[inu11 + 1];
  const float* Wn12 = (const float*)d_in[inu12], *bn12 = (const float*)d_in[inu12 + 1];
  const float* Wn13 = (const float*)d_in[inu13], *bn13 = (const float*)d_in[inu13 + 1];
  const float* Wed1 = (const float*)d_in[ied1],  *bed1 = (const float*)d_in[ied1 + 1];
  const float* Weu1 = (const float*)d_in[ieu1],  *beu1 = (const float*)d_in[ieu1 + 1];
  const float* Wnb2 = (const float*)d_in[inb2],  *bnb2 = (const float*)d_in[inb2 + 1];
  const float* Wn21 = (const float*)d_in[inu21], *bn21 = (const float*)d_in[inu21 + 1];
  const float* Wn22 = (const float*)d_in[inu22], *bn22 = (const float*)d_in[inu22 + 1];
  const float* Wn23 = (const float*)d_in[inu23], *bn23 = (const float*)d_in[inu23 + 1];
  const float* Wfc  = (const float*)d_in[ifc],   *bfc  = (const float*)d_in[ifc + 1];

  char* base = (char*)d_ws; size_t off = 0;
  auto alloc = [&](size_t bytes) -> void* {
    off = (off + 255) & ~(size_t)255; void* p = base + off; off += bytes; return p;
  };
  bf16*  xp   = (bf16*)alloc((size_t)N * 32 * 2);
  bf16*  eap  = (bf16*)alloc((size_t)E * 32 * 2);
  float* ss1  = (float*)alloc((size_t)N * 64 * 4);
  int*   cnt  = (int*)alloc((size_t)N * 4);
  bf16*  nin1 = (bf16*)alloc((size_t)N * 64 * 2);
  bf16*  t1   = (bf16*)alloc((size_t)N * 96 * 2);
  bf16*  t2   = (bf16*)alloc((size_t)N * 96 * 2);
  bf16*  h1   = (bf16*)alloc((size_t)N * 128 * 2);
  float* ss2  = (float*)alloc((size_t)N * 128 * 4);
  bf16*  nin2 = (bf16*)alloc((size_t)N * 256 * 2);
  bf16*  u1   = (bf16*)alloc((size_t)N * 192 * 2);
  bf16*  u2   = (bf16*)alloc((size_t)N * 96 * 2);
  bf16*  h2   = (bf16*)alloc((size_t)N * 288 * 2);
  float* Wf   = (float*)alloc(7 * 128 * 4);     // Wed @ Weu[256:384]
  float* be1  = (float*)alloc(128 * 4);         // folded e1 bias
  float* S0   = (float*)alloc(128 * 112 * 4);   // Wnb2[0:128] + Weu[0:128]@Wnb2[128:256]
  float* S1   = (float*)alloc(128 * 112 * 4);   // Weu[128:256]@Wnb2[128:256]
  float* S2   = (float*)alloc(7 * 112 * 4);     // Wf@Wnb2[128:256]
  float* bm2  = (float*)alloc(112 * 4);         // folded msg2 bias
  bf16* Wnb1x = (bf16*)alloc(64 * 32 * 2);
  bf16* Wnb1e = (bf16*)alloc(64 * 32 * 2);
  bf16* Wt11  = (bf16*)alloc(96 * 64 * 2);
  bf16* Wt12  = (bf16*)alloc(96 * 96 * 2);
  bf16* Wt13  = (bf16*)alloc(128 * 96 * 2);
  bf16* WtS0  = (bf16*)alloc(128 * 128 * 2);
  bf16* WtS1  = (bf16*)alloc(128 * 128 * 2);
  bf16* WtS2  = (bf16*)alloc(128 * 32 * 2);
  bf16* Wt21  = (bf16*)alloc(192 * 256 * 2);
  bf16* Wt22  = (bf16*)alloc(96 * 192 * 2);
  bf16* Wt23  = (bf16*)alloc(288 * 96 * 2);
  if (off > ws_size) return;

  auto g1 = [](long long n) { return dim3((unsigned)((n + 255) / 256)); };

  // --- stage padded bf16 activations of the raw inputs
  pad3_k<<<g1((long long)N * 32), 256, 0, stream>>>(x, xp, N);
  pad7_k<<<g1((long long)E * 32), 256, 0, stream>>>(ea, eap, E);

  // --- on-device weight fusion (e_t and e1 folded away) + bf16 transpose/pad
  mm_k<<<g1(7 * 128), 256, 0, stream>>>(Wf, Wed1, 128, 0, Weu1, 128, 256, nullptr, 0, 0, 7, 128, 128);
  mm_k<<<g1(128), 256, 0, stream>>>(be1, bed1, 128, 0, Weu1, 128, 256, beu1, 128, 0, 1, 128, 128);
  mm_k<<<g1(128 * 112), 256, 0, stream>>>(S0, Weu1, 128, 0, Wnb2, 112, 128, Wnb2, 112, 0, 128, 128, 112);
  mm_k<<<g1(128 * 112), 256, 0, stream>>>(S1, Weu1, 128, 128, Wnb2, 112, 128, nullptr, 0, 0, 128, 128, 112);
  mm_k<<<g1(7 * 112), 256, 0, stream>>>(S2, Wf, 128, 0, Wnb2, 112, 128, nullptr, 0, 0, 7, 128, 112);
  mm_k<<<g1(112), 256, 0, stream>>>(bm2, be1, 128, 0, Wnb2, 112, 128, bnb2, 112, 0, 1, 128, 112);

  cvt_k<<<g1(64 * 32), 256, 0, stream>>>(Wnb1, 42, 0, 3, 42, Wnb1x, 32, 64);
  cvt_k<<<g1(64 * 32), 256, 0, stream>>>(Wnb1, 42, 3, 7, 42, Wnb1e, 32, 64);
  cvt_k<<<g1(96 * 64), 256, 0, stream>>>(Wn11, 83, 0, 45, 83, Wt11, 64, 96);
  cvt_k<<<g1(96 * 96), 256, 0, stream>>>(Wn12, 71, 0, 83, 71, Wt12, 96, 96);
  cvt_k<<<g1(128 * 96), 256, 0, stream>>>(Wn13, 128, 0, 71, 128, Wt13, 96, 128);
  cvt_k<<<g1(128 * 128), 256, 0, stream>>>(S0, 112, 0, 128, 112, WtS0, 128, 128);
  cvt_k<<<g1(128 * 128), 256, 0, stream>>>(S1, 112, 0, 128, 112, WtS1, 128, 128);
  cvt_k<<<g1(128 * 32), 256, 0, stream>>>(S2, 112, 0, 7, 112, WtS2, 32, 128);
  cvt_k<<<g1(192 * 256), 256, 0, stream>>>(Wn21, 184, 0, 240, 184, Wt21, 256, 192);
  cvt_k<<<g1(96 * 192), 256, 0, stream>>>(Wn22, 80, 0, 184, 80, Wt22, 192, 96);
  cvt_k<<<g1(288 * 96), 256, 0, stream>>>(Wn23, 264, 0, 80, 264, Wt23, 96, 288);

  // --- layer 1: edge msg -> mean aggregate -> node MLP
  zero_k<<<g1((long long)N * 64), 256, 0, stream>>>(ss1, (long long)N * 64);
  zero_k<<<g1(N), 256, 0, stream>>>((float*)cnt, N);
  count_k<<<g1(E), 256, 0, stream>>>(col, cnt, E);
  gemm(stream, E, 64, 42, bnb1,
       mkseg(xp, row, Wnb1x, 32), mkseg(eap, nullptr, Wnb1e, 32), mkseg(nullptr, nullptr, nullptr, 0), 2,
       nullptr, 0, 0, ss1, col, 64);
  finish1_k<<<g1((long long)N * 64), 256, 0, stream>>>(x, ss1, cnt, nin1, N);
  gemm(stream, N, 96, 83, bn11, mkseg(nin1, nullptr, Wt11, 64),
       mkseg(nullptr, nullptr, nullptr, 0), mkseg(nullptr, nullptr, nullptr, 0), 1,
       t1, 96, 1, nullptr, nullptr, 0);
  gemm(stream, N, 96, 71, bn12, mkseg(t1, nullptr, Wt12, 96),
       mkseg(nullptr, nullptr, nullptr, 0), mkseg(nullptr, nullptr, nullptr, 0), 1,
       t2, 96, 1, nullptr, nullptr, 0);
  gemm(stream, N, 128, 128, bn13, mkseg(t2, nullptr, Wt13, 96),
       mkseg(nullptr, nullptr, nullptr, 0), mkseg(nullptr, nullptr, nullptr, 0), 1,
       h1, 128, 0, nullptr, nullptr, 0);

  // --- layer 2: fused (eu1 -> nb2) edge msg -> mean aggregate -> node MLP
  zero_k<<<g1((long long)N * 128), 256, 0, stream>>>(ss2, (long long)N * 128);
  gemm(stream, E, 128, 112, bm2,
       mkseg(h1, row, WtS0, 128), mkseg(h1, col, WtS1, 128), mkseg(eap, nullptr, WtS2, 32), 3,
       nullptr, 0, 0, ss2, col, 128);
  finish2_k<<<g1((long long)N * 256), 256, 0, stream>>>(h1, ss2, cnt, nin2, N);
  gemm(stream, N, 192, 184, bn21, mkseg(nin2, nullptr, Wt21, 256),
       mkseg(nullptr, nullptr, nullptr, 0), mkseg(nullptr, nullptr, nullptr, 0), 1,
       u1, 192, 1, nullptr, nullptr, 0);
  gemm(stream, N, 96, 80, bn22, mkseg(u1, nullptr, Wt22, 192),
       mkseg(nullptr, nullptr, nullptr, 0), mkseg(nullptr, nullptr, nullptr, 0), 1,
       u2, 96, 1, nullptr, nullptr, 0);
  gemm(stream, N, 288, 264, bn23, mkseg(u2, nullptr, Wt23, 96),
       mkseg(nullptr, nullptr, nullptr, 0), mkseg(nullptr, nullptr, nullptr, 0), 1,
       h2, 288, 0, nullptr, nullptr, 0);

  // --- fc predict
  fc_k<<<g1(N), 256, 0, stream>>>(h2, Wfc, bfc, (float*)d_out, N);
}
